// Seq2SeqLSTMAttention_51041391345689
// MI455X (gfx1250) — compile-verified, matrix operand order checked by reading, not run on hardware
//
#include <hip/hip_runtime.h>
#include <hip/hip_bf16.h>

// ---------------------------------------------------------------------------
// Luong "general" attention for MI455X (gfx1250, wave32, WMMA).
// All GEMMs: bf16 inputs (converted on the fly), fp32 accumulate via
// v_wmma_f32_16x16x32_bf16. Softmax / bias / mask / tanh epilogues in fp32.
// Pure-copy bf16 operand streams use GLOBAL_LOAD_ASYNC_TO_LDS_B128
// (ASYNCcnt + s_wait_asynccnt fence before the workgroup barrier).
// ---------------------------------------------------------------------------

typedef __attribute__((ext_vector_type(16))) __bf16 v16bf;
typedef __attribute__((ext_vector_type(8)))  __bf16 bf16x8;
typedef __attribute__((ext_vector_type(4)))  __bf16 bf16x4;
typedef __attribute__((ext_vector_type(2)))  __bf16 bf16x2;
typedef __attribute__((ext_vector_type(8)))  float  v8f;
typedef int v4i __attribute__((vector_size(16)));

#define AS1 __attribute__((address_space(1)))
#define AS3 __attribute__((address_space(3)))

static constexpr float kNegInf = -1.0e10f;

// Problem sizes (fixed by the reference)
static constexpr int Bn  = 64;
static constexpr int Tn  = 128;
static constexpr int Sn  = 512;
static constexpr int ENC = 1024;
static constexpr int DEC = 512;

// Block tiling
static constexpr int BM = 128, BN = 128, BK = 32;
static constexpr int LDSK = 40; // padded K-stride (bf16 elems): 80B rows, 16B aligned

#if defined(__gfx1250__) && __has_builtin(__builtin_amdgcn_global_load_async_to_lds_b128)
#define HAVE_ASYNC_LDS 1
#else
#define HAVE_ASYNC_LDS 0
#endif

__device__ __forceinline__ void async_copy_b128(const void* gp, void* lp) {
#if HAVE_ASYNC_LDS
  AS1 v4i* g = (AS1 v4i*)(AS1 void*)const_cast<void*>(gp);
  AS3 v4i* l = (AS3 v4i*)(AS3 void*)lp;
  __builtin_amdgcn_global_load_async_to_lds_b128(g, l, 0, 0);
#else
  *(bf16x8*)lp = *(const bf16x8*)gp;
#endif
}

__device__ __forceinline__ void wait_async() {
#if HAVE_ASYNC_LDS
#if __has_builtin(__builtin_amdgcn_s_wait_asynccnt)
  __builtin_amdgcn_s_wait_asynccnt(0);
#else
  asm volatile("s_wait_asynccnt 0x0" ::: "memory");
#endif
#endif
}

// Load one 16x32 bf16 operand fragment from a K-major LDS tile.
// ISA layout (CDNA5 ISA): lanes 0-15 hold K=0..7 (VGPR0-3) and K=16..23
// (VGPR4-7); lanes 16-31 hold K=8..15 and K=24..31. rowp points at the
// fragment's row (M or N index already applied).
__device__ __forceinline__ v16bf load_frag(const __bf16* rowp, int half) {
  union { v16bf v; bf16x8 h[2]; } u;
  u.h[0] = *(const bf16x8*)(rowp + half * 8);
  u.h[1] = *(const bf16x8*)(rowp + 16 + half * 8);
  return u.v;
}

// EPI: 0 = (acc+bias[n])*mask[row] -> bf16   (proj)
//      1 = acc + NEG_INF*(1-mask[z,n]) -> f32 (masked energies)
//      2 = acc -> bf16                        (context into concat buffer)
//      3 = tanh(acc) -> f32                   (h_tilde)
// A_BF16: A operand already bf16 in global memory (else f32, convert on load)
// B_NK:   B operand stored [N][K] bf16 (direct copy); else f32 [K][N]
//         (transpose + convert while staging to LDS)
template <int EPI, bool A_BF16, bool B_NK>
__global__ __launch_bounds__(256) void wmma_gemm(
    const void* __restrict__ Ag, int lda, long long sA,
    const void* __restrict__ Bg, int ldb, long long sB,
    void* __restrict__ Og, int ldc, long long sO,
    const float* __restrict__ bias,
    const int* __restrict__ mask, long long maskStride,
    int K) {
  __shared__ __bf16 As[BM * LDSK];
  __shared__ __bf16 Bs[BN * LDSK];

  const int tid  = threadIdx.x;
  const int wave = tid >> 5;
  const int lane = tid & 31;
  const int r    = lane & 15;
  const int half = lane >> 4;
  const int wm   = wave >> 2;   // 0..1  (64-row wave tile)
  const int wn   = wave & 3;    // 0..3  (32-col wave tile)
  const int z    = blockIdx.z;
  const long long blockM = (long long)blockIdx.y * BM;
  const long long blockN = (long long)blockIdx.x * BN;

  const float*  Af = (const float*)Ag;
  const __bf16* Ab = (const __bf16*)Ag;
  const float*  Bf = (const float*)Bg;
  const __bf16* Bb = (const __bf16*)Bg;

  v8f acc[4][2] = {};

  const int nK = K / BK;
  for (int kt = 0; kt < nK; ++kt) {
    const int kbase = kt * BK;

    // ---- stage A tile (BM x BK), K-major, convert f32->bf16 if needed ----
    if constexpr (!A_BF16) {
#pragma unroll
      for (int i = 0; i < 4; ++i) {
        int f   = i * 256 + tid;          // 1024 float4 chunks
        int row = f >> 3;
        int col = (f & 7) * 4;
        const float* gp = Af + z * sA + (blockM + row) * lda + kbase + col;
        const float4 v = *(const float4*)gp;
        if (i == 0 && kt + 1 < nK) __builtin_prefetch(gp + BK, 0, 1);
        bf16x4 o;
        o.x = (__bf16)v.x; o.y = (__bf16)v.y; o.z = (__bf16)v.z; o.w = (__bf16)v.w;
        *(bf16x4*)(As + row * LDSK + col) = o;
      }
    } else {
#pragma unroll
      for (int i = 0; i < 2; ++i) {
        int f   = i * 256 + tid;          // 512 bf16x8 chunks
        int row = f >> 2;
        int col = (f & 3) * 8;
        async_copy_b128(Ab + z * sA + (blockM + row) * lda + kbase + col,
                        As + row * LDSK + col);
      }
    }

    // ---- stage B tile as Bt[n][k] (BN x BK) ----
    if constexpr (B_NK) {
#pragma unroll
      for (int i = 0; i < 2; ++i) {
        int f = i * 256 + tid;
        int n = f >> 2;
        int c = (f & 3) * 8;
        async_copy_b128(Bb + z * sB + (blockN + n) * ldb + kbase + c,
                        Bs + n * LDSK + c);
      }
    } else {
      // f32 [K][N]: each thread handles a 2(K) x 4(N) patch so the transposed
      // LDS stores become packed 32-bit (k,k+1) pairs -> ds_store_b32.
#pragma unroll
      for (int i = 0; i < 2; ++i) {
        int p = i * 256 + tid;            // 512 patches
        int k = (p >> 5) * 2;
        int n = (p & 31) * 4;
        const float* gp0 =
            Bf + z * sB + (long long)(kbase + k) * ldb + blockN + n;
        const float* gp1 = gp0 + ldb;
        const float4 va = *(const float4*)gp0;
        const float4 vb = *(const float4*)gp1;
        if (i == 0 && kt + 1 < nK) __builtin_prefetch(gp0 + (long long)BK * ldb, 0, 1);
        bf16x2 t0, t1, t2, t3;
        t0.x = (__bf16)va.x; t0.y = (__bf16)vb.x;
        t1.x = (__bf16)va.y; t1.y = (__bf16)vb.y;
        t2.x = (__bf16)va.z; t2.y = (__bf16)vb.z;
        t3.x = (__bf16)va.w; t3.y = (__bf16)vb.w;
        *(bf16x2*)(Bs + (n + 0) * LDSK + k) = t0;
        *(bf16x2*)(Bs + (n + 1) * LDSK + k) = t1;
        *(bf16x2*)(Bs + (n + 2) * LDSK + k) = t2;
        *(bf16x2*)(Bs + (n + 3) * LDSK + k) = t3;
      }
    }

    // Each wave drains its own ASYNCcnt before signaling, so the barrier is a
    // valid completion fence for every wave's async-to-LDS data.
    if constexpr (A_BF16 || B_NK) wait_async();
    __syncthreads();

    // ---- one K=32 WMMA step over the 64x32 wave tile ----
    v16bf af[4], bfr[2];
#pragma unroll
    for (int ti = 0; ti < 4; ++ti)
      af[ti] = load_frag(As + (wm * 64 + ti * 16 + r) * LDSK, half);
#pragma unroll
    for (int tj = 0; tj < 2; ++tj)
      bfr[tj] = load_frag(Bs + (wn * 32 + tj * 16 + r) * LDSK, half);

#pragma unroll
    for (int ti = 0; ti < 4; ++ti)
#pragma unroll
      for (int tj = 0; tj < 2; ++tj)
        acc[ti][tj] = __builtin_amdgcn_wmma_f32_16x16x32_bf16(
            false, af[ti], false, bfr[tj], (short)0, acc[ti][tj], false, false);

    __syncthreads();
  }

  // ---- epilogue. C layout: VGPR v holds M=v (lanes 0-15) / M=v+8 (16-31),
  //      N = lane&15. ----
#pragma unroll
  for (int ti = 0; ti < 4; ++ti) {
#pragma unroll
    for (int tj = 0; tj < 2; ++tj) {
      const long long gm0 = blockM + wm * 64 + ti * 16 + 8 * half;
      const long long gn  = blockN + wn * 32 + tj * 16 + r;
      v8f c = acc[ti][tj];
      if constexpr (EPI == 0) {
        const float bv = bias[gn];
        __bf16* O = (__bf16*)Og;
#pragma unroll
        for (int v = 0; v < 8; ++v) {
          const long long row = gm0 + v;
          const float mv = (float)mask[row];
          O[row * ldc + gn] = (__bf16)((c[v] + bv) * mv);
        }
      } else if constexpr (EPI == 1) {
        const float add = mask[z * maskStride + gn] ? 0.0f : kNegInf;
        float* O = (float*)Og + z * sO;
#pragma unroll
        for (int v = 0; v < 8; ++v) O[(gm0 + v) * ldc + gn] = c[v] + add;
      } else if constexpr (EPI == 2) {
        __bf16* O = (__bf16*)Og + z * sO;
#pragma unroll
        for (int v = 0; v < 8; ++v) O[(gm0 + v) * ldc + gn] = (__bf16)c[v];
      } else {
        float* O = (float*)Og + z * sO;
#pragma unroll
        for (int v = 0; v < 8; ++v) O[(gm0 + v) * ldc + gn] = tanhf(c[v]);
      }
    }
  }
}

// Row softmax over S=512: one block per (b,t) row; writes f32 attention and a
// bf16 copy for the context GEMM.
__global__ __launch_bounds__(256) void softmax_rows(
    const float* __restrict__ me, float* __restrict__ attnF,
    __bf16* __restrict__ attnB) {
  const int row = blockIdx.x;
  const int t   = threadIdx.x;
  const float* x = me + (long long)row * Sn;
  const float a = x[t];
  const float b = x[t + 256];

  __shared__ float red[256];
  red[t] = fmaxf(a, b);
  __syncthreads();
  for (int s = 128; s > 0; s >>= 1) {
    if (t < s) red[t] = fmaxf(red[t], red[t + s]);
    __syncthreads();
  }
  const float mx = red[0];
  __syncthreads();

  const float e0 = expf(a - mx);
  const float e1 = expf(b - mx);
  red[t] = e0 + e1;
  __syncthreads();
  for (int s = 128; s > 0; s >>= 1) {
    if (t < s) red[t] += red[t + s];
    __syncthreads();
  }
  const float inv = 1.0f / red[0];

  const float w0 = e0 * inv, w1 = e1 * inv;
  const long long o = (long long)row * Sn;
  attnF[o + t]        = w0;
  attnF[o + t + 256]  = w1;
  attnB[o + t]        = (__bf16)w0;
  attnB[o + t + 256]  = (__bf16)w1;
}

// hcat[:, :, 1024:1536] = bf16(dec_hidden)
__global__ __launch_bounds__(256) void fill_dec(const float* __restrict__ dec,
                                                __bf16* __restrict__ hcat) {
  const long long idx = (long long)blockIdx.x * 256 + threadIdx.x;
  const long long row = idx >> 9;     // /512
  const int d = (int)(idx & 511);
  hcat[row * (ENC + DEC) + ENC + d] = (__bf16)dec[idx];
}

extern "C" void kernel_launch(void* const* d_in, const int* in_sizes, int n_in,
                              void* d_out, int out_size, void* d_ws,
                              size_t ws_size, hipStream_t stream) {
  const float* dec   = (const float*)d_in[0];  // [B,T,DEC]
  const float* enc   = (const float*)d_in[1];  // [B,S,ENC]
  const int*   emask = (const int*)d_in[2];    // [B,S]
  const float* Wa    = (const float*)d_in[3];  // [ENC,DEC]
  const float* ba    = (const float*)d_in[4];  // [DEC]
  const float* Wo    = (const float*)d_in[5];  // [ENC+DEC,DEC]

  float* out     = (float*)d_out;
  float* h_tilde = out;                                   // [B,T,DEC]
  float* attnF   = out + (long long)Bn * Tn * DEC;        // [B,T,S]
  float* menerg  = attnF + (long long)Bn * Tn * Sn;       // [B,T,S]

  char* ws = (char*)d_ws;
  __bf16* proj  = (__bf16*)ws;                            // [B,S,DEC]   32 MB
  __bf16* attnB = (__bf16*)(ws + 33554432);               // [B,T,S]      8 MB
  __bf16* hcat  = (__bf16*)(ws + 41943040);               // [B,T,1536]  24 MB

  dim3 blk(256);

  // 1) proj = (enc @ W_attn + b) * mask   [32768,1024]x[1024,512] -> bf16
  wmma_gemm<0, false, false><<<dim3(DEC / BN, (Bn * Sn) / BM, 1), blk, 0,
                               stream>>>(
      enc, ENC, 0, Wa, DEC, 0, proj, DEC, 0, ba, emask, 0, ENC);

  // 2) masked_energies = dec @ proj^T + NEG_INF*(1-mask)  (batched over B)
  wmma_gemm<1, false, true><<<dim3(Sn / BN, 1, Bn), blk, 0, stream>>>(
      dec, DEC, (long long)Tn * DEC, proj, DEC, (long long)Sn * DEC, menerg,
      Sn, (long long)Tn * Sn, nullptr, emask, Sn, DEC);

  // 3) softmax rows -> attn (f32 output + bf16 copy)
  softmax_rows<<<dim3(Bn * Tn), blk, 0, stream>>>(menerg, attnF, attnB);

  // 4a) concat buffer: dec half
  fill_dec<<<dim3((Bn * Tn * DEC) / 256), blk, 0, stream>>>(dec, hcat);

  // 4b) context = attn @ enc  (batched) -> bf16 into hcat[:, :1024]
  wmma_gemm<2, true, false><<<dim3(ENC / BN, 1, Bn), blk, 0, stream>>>(
      attnB, Sn, (long long)Tn * Sn, enc, ENC, (long long)Sn * ENC, hcat,
      ENC + DEC, (long long)Tn * (ENC + DEC), nullptr, nullptr, 0, Sn);

  // 5) h_tilde = tanh(hcat @ W_out)   [8192,1536]x[1536,512] -> f32
  wmma_gemm<3, true, false><<<dim3(DEC / BN, (Bn * Tn) / BM, 1), blk, 0,
                              stream>>>(
      hcat, ENC + DEC, 0, Wo, DEC, 0, h_tilde, DEC, 0, nullptr, nullptr, 0,
      ENC + DEC);
}
